// RNNModel_28475633172675
// MI455X (gfx1250) — matrix operand order repaired and checked by reading, not compiled, not run on hardware
//
#include <hip/hip_runtime.h>
#include <math.h>

// Problem dims
#define B_  64
#define T_  512
#define V_  50000
#define E_  300
#define EP_ 320      // E padded to multiple of 32 (WMMA K step)
#define I_  512
#define H_  1024

typedef __bf16 bf16_t;
typedef __attribute__((ext_vector_type(16))) __bf16 v16bf;
typedef __attribute__((ext_vector_type(8)))  __bf16 v8bf;
typedef __attribute__((ext_vector_type(8)))  float  v8f;
typedef __attribute__((ext_vector_type(4)))  unsigned int v4u;
typedef __attribute__((ext_vector_type(8)))  int v8i;
typedef __attribute__((ext_vector_type(4)))  int v4i;
typedef unsigned short u16;
typedef unsigned int   u32;
typedef unsigned long long u64;

// float -> bf16 round-to-nearest-even, stored as raw u16
__device__ __forceinline__ u16 f2bf(float f) {
    u32 u = __builtin_bit_cast(u32, f);
    u32 r = u + 0x7FFFu + ((u >> 16) & 1u);
    return (u16)(r >> 16);
}

union Frag16 { v16bf v; v8bf h[2]; };

// A fragment: 16x32 bf16, row-major source (lda elements/row).
// lane l<16: row l, K chunks [k0,k0+8) and [k0+16,k0+24); lanes 16-31: +8 shift.
__device__ __forceinline__ v16bf load_a_frag(const u16* A, int lda, int m0, int k0, int lane) {
    const int r  = lane & 15;
    const int hi = lane >> 4;
    const u16* p = A + (size_t)(m0 + r) * lda + k0 + hi * 8;
    Frag16 f;
    f.h[0] = *(const v8bf*)(p);
    f.h[1] = *(const v8bf*)(p + 16);
    return f.v;
}

// B fragment: 32x16 bf16 where B[k][n] = W[n0+n][k], W row-major K-contiguous.
// lane l: column n=l&15; K chunks [k0,k0+8) and [k0+16,k0+24); lanes 16-31: +8 shift.
__device__ __forceinline__ v16bf load_b_frag(const u16* W, int ldb, int n0, int k0, int lane) {
    const int n  = lane & 15;
    const int hi = lane >> 4;
    const u16* p = W + (size_t)(n0 + n) * ldb + k0 + hi * 8;
    Frag16 f;
    f.h[0] = *(const v8bf*)(p);
    f.h[1] = *(const v8bf*)(p + 16);
    return f.v;
}

__device__ __forceinline__ v8f wmma_bf16(v16bf a, v16bf b, v8f c) {
    return __builtin_amdgcn_wmma_f32_16x16x32_bf16(
        /*neg_a=*/false, a, /*neg_b=*/false, b,
        /*c_mod=*/(short)0, c, /*reuse_a=*/false, /*reuse_b=*/false);
}

// ---------------------------------------------------------------------------
// K0a: W_comb[h,e] = sum_i W_ih[h,i]*W_proj[i,e]  (bf16, padded to EP_)
//      bias[h]     = b_ih[h] + b_hh[h] + sum_i W_ih[h,i]*b_proj[i]
__global__ void prep_weights_kernel(const float* __restrict__ W_ih,
                                    const float* __restrict__ W_proj,
                                    const float* __restrict__ b_proj,
                                    const float* __restrict__ b_ih,
                                    const float* __restrict__ b_hh,
                                    u16* __restrict__ Wc, float* __restrict__ bias) {
    const int h = blockIdx.x;
    const int e = threadIdx.x;
    float s = 0.f;
    if (e < E_) {
        const float* wrow = W_ih + (size_t)h * I_;
        for (int i = 0; i < I_; ++i) s += wrow[i] * W_proj[(size_t)i * E_ + e];
    }
    Wc[(size_t)h * EP_ + e] = (e < E_) ? f2bf(s) : (u16)0;
    if (e == 0) {
        const float* wrow = W_ih + (size_t)h * I_;
        float sb = b_ih[h] + b_hh[h];
        for (int i = 0; i < I_; ++i) sb += wrow[i] * b_proj[i];
        bias[h] = sb;
    }
}

// K0b: W_hh f32 -> bf16 (row-major kept: B columns == W_hh rows)
__global__ void conv_whh_kernel(const float* __restrict__ W, u16* __restrict__ Wb) {
    const int idx = blockIdx.x * blockDim.x + threadIdx.x;
    Wb[idx] = f2bf(W[idx]);
}

// K1: gather + convert: Xg[m, 0:300] = bf16(emb[x[m], :]), zero-pad to 320
__global__ void gather_embed_kernel(const int* __restrict__ x,
                                    const float* __restrict__ emb,
                                    u16* __restrict__ Xg) {
    const int m = blockIdx.x;
    const int e = threadIdx.x;
    const int tok = x[m];
    Xg[(size_t)m * EP_ + e] = (e < E_) ? f2bf(emb[(size_t)tok * E_ + e]) : (u16)0;
}

// K2: pre (32768 x 1024) = Xg (32768 x 320) @ Wc^T + bias   [WMMA bf16]
// Each wave computes a 16x64 strip (4 accumulators sharing one A fragment).
__global__ void gemm_pre_kernel(const u16* __restrict__ Xg, const u16* __restrict__ Wc,
                                const float* __restrict__ bias, float* __restrict__ pre) {
    const int lane = threadIdx.x & 31;
    const int gw   = blockIdx.x * 8 + (threadIdx.x >> 5);
    const int m0   = (gw >> 4) * 16;
    const int n0   = (gw & 15) * 64;
    const int r    = lane & 15;
    const int hi   = lane >> 4;

    v8f acc[4];
    #pragma unroll
    for (int j = 0; j < 4; ++j) {
        const float bj = bias[n0 + j * 16 + r];
        #pragma unroll
        for (int v = 0; v < 8; ++v) acc[j][v] = bj;
    }
    for (int k0 = 0; k0 < EP_; k0 += 32) {
        const v16bf a = load_a_frag(Xg, EP_, m0, k0, lane);
        #pragma unroll
        for (int j = 0; j < 4; ++j) {
            const v16bf b = load_b_frag(Wc, EP_, n0 + j * 16, k0, lane);
            acc[j] = wmma_bf16(a, b, acc[j]);
        }
    }
    #pragma unroll
    for (int j = 0; j < 4; ++j)
        #pragma unroll
        for (int v = 0; v < 8; ++v)
            pre[(size_t)(m0 + v + 8 * hi) * H_ + n0 + j * 16 + r] = acc[j][v];
}

// ---------------------------------------------------------------------------
// K3: persistent recurrence: for t in 0..511: h = tanh(pre_t + h @ W_hh^T)
// 8 blocks x 8 waves. Block bb owns state columns [bb*128, bb*128+128).
// Its 128x1024 bf16 slice of W_hh (256 KB) is DMA'd into LDS ONCE via the
// Tensor Data Mover, then all 512 steps read B fragments from LDS.
// Wave w: mt = w>>1 (16-row tile), n-half = w&1 (64 cols), 4 accumulators.
// Grid barrier via device-scope atomic counter between steps.
__global__ void rnn_scan_kernel(const float* __restrict__ pre, const u16* __restrict__ Whh,
                                u16* __restrict__ h0, u16* __restrict__ h1,
                                float* __restrict__ hT, u32* __restrict__ ctr) {
    extern __shared__ u16 Wlds[];   // [128][1024] bf16 = 256 KB

    const int lane = threadIdx.x & 31;
    const int wave = threadIdx.x >> 5;          // 0..7
    const int bb   = blockIdx.x;                // 0..7
    const int m0   = (wave >> 1) * 16;          // 4 M-tiles of the 64-row state
    const int nl0  = (wave & 1) * 64;           // column offset within block slice
    const int n0   = bb * 128 + nl0;            // global column offset
    const int r    = lane & 15;
    const int hi   = lane >> 4;

    // ---- One-shot TDM: LDS[0:128][0:1024] = Whh[bb*128 : bb*128+128][:] ----
    if (wave == 0) {
        const u64 ga  = (u64)(const void*)(Whh + (size_t)bb * 128 * H_);
        const u32 lba = (u32)(size_t)(void*)Wlds;   // LDS aperture: low 32 bits = LDS byte offset
        // D# group 0: count=1 | lds_addr | global_addr[56:0] | type=2
        v4u g0;
        g0[0] = 1u;                                  // count=1 (valid descriptor)
        g0[1] = lba;                                 // lds_addr
        g0[2] = (u32)ga;                             // global_addr[31:0]
        g0[3] = (u32)((ga >> 32) & 0x01FFFFFFu) | (2u << 30);  // global_addr[56:32] | type=2
        // D# group 1: 2D tile, data_size=2B, tile 1024x128 of a 1024-stride tensor
        v8i g1;
        g1[0] = (int)(1u << 16);       // workgroup_mask=0, data_size=1 (2 bytes)
        g1[1] = (int)(1024u << 16);    // atomic_barrier_addr=0, tensor_dim0[15:0]=1024
        g1[2] = (int)(1024u << 16);    // tensor_dim0[31:16]=0, tensor_dim1[15:0]=1024
        g1[3] = (int)(1024u << 16);    // tensor_dim1[31:16]=0, tile_dim0=1024
        g1[4] = (int)128;              // tile_dim1=128, tile_dim2=0
        g1[5] = (int)1024;             // tensor_dim0_stride[31:0]=1024 elements
        g1[6] = 0;                     // tensor_dim0_stride[47:32]=0, tensor_dim1_stride lo=0
        g1[7] = 0;                     // tensor_dim1_stride hi=0
        v4i gz4 = { 0, 0, 0, 0 };                    // groups 2/3 unused (2D tensor)
        v8i gz8 = { 0, 0, 0, 0, 0, 0, 0, 0 };        // extra group (unused)
        __builtin_amdgcn_tensor_load_to_lds(g0, g1, gz4, gz4, gz8, 0);
        __builtin_amdgcn_s_wait_tensorcnt(0);
    }
    __syncthreads();

    for (int t = 0; t < T_; ++t) {
        const u16* hc = (t & 1) ? h1 : h0;
        u16*       hn = (t & 1) ? h0 : h1;

        // prefetch next timestep's pre tile while we compute this one
        if (t + 1 < T_)
            __builtin_prefetch(&pre[((size_t)(m0 + r) * T_ + (t + 1)) * H_ + n0], 0, 1);

        v8f acc[4];
        #pragma unroll
        for (int j = 0; j < 4; ++j)
            #pragma unroll
            for (int v = 0; v < 8; ++v)
                acc[j][v] = pre[((size_t)(m0 + v + 8 * hi) * T_ + t) * H_ + n0 + j * 16 + r];

        for (int k0 = 0; k0 < H_; k0 += 32) {
            const v16bf a = load_a_frag(hc, H_, m0, k0, lane);
            #pragma unroll
            for (int j = 0; j < 4; ++j) {
                // B fragment from LDS: local row = nl0 + j*16 + (lane&15)
                const v16bf b = load_b_frag(Wlds, H_, nl0 + j * 16, k0, lane);
                acc[j] = wmma_bf16(a, b, acc[j]);
            }
        }

        #pragma unroll
        for (int j = 0; j < 4; ++j)
            #pragma unroll
            for (int v = 0; v < 8; ++v) {
                const int m = m0 + v + 8 * hi;
                const int n = n0 + j * 16 + r;
                const float y = tanhf(acc[j][v]);
                hn[(size_t)m * H_ + n] = f2bf(y);
                if (t == T_ - 1) hT[(size_t)m * H_ + n] = y;
            }

        // device-scope grid barrier (monotonic counter, 8 workgroups)
        __syncthreads();
        if (threadIdx.x == 0) {
            __hip_atomic_fetch_add(ctr, 1u, __ATOMIC_RELEASE, __HIP_MEMORY_SCOPE_AGENT);
            const u32 target = 8u * (u32)(t + 1);
            while (__hip_atomic_load(ctr, __ATOMIC_ACQUIRE, __HIP_MEMORY_SCOPE_AGENT) < target) {
                __builtin_amdgcn_s_sleep(1);
            }
        }
        __syncthreads();
    }
}

// K4: out[b] = dot(hT[b,:], W_fc) + b_fc  -- 1 block, 1024 threads (16 lanes per b)
__global__ void fc_out_kernel(const float* __restrict__ hT, const float* __restrict__ W_fc,
                              const float* __restrict__ b_fc, float* __restrict__ out) {
    const int tid = threadIdx.x;
    const int b = tid >> 4;
    const int l = tid & 15;
    float s = 0.f;
    for (int h = l; h < H_; h += 16) s += hT[(size_t)b * H_ + h] * W_fc[h];
    for (int off = 8; off; off >>= 1) s += __shfl_down(s, off, 16);
    if (l == 0) out[b] = s + b_fc[0];
}

extern "C" void kernel_launch(void* const* d_in, const int* in_sizes, int n_in,
                              void* d_out, int out_size, void* d_ws, size_t ws_size,
                              hipStream_t stream) {
    const int*   x      = (const int*)  d_in[0];
    const float* emb    = (const float*)d_in[1];
    const float* W_proj = (const float*)d_in[2];
    const float* b_proj = (const float*)d_in[3];
    const float* W_ih   = (const float*)d_in[4];
    const float* b_ih   = (const float*)d_in[5];
    const float* W_hh   = (const float*)d_in[6];
    const float* b_hh   = (const float*)d_in[7];
    const float* W_fc   = (const float*)d_in[8];
    const float* b_fc   = (const float*)d_in[9];
    float* out = (float*)d_out;

    char* ws = (char*)d_ws;
    size_t o = 0;
    u16*   Xg   = (u16*)  (ws + o); o += (size_t)B_ * T_ * EP_ * 2;   // 20.0 MB
    u16*   Wc   = (u16*)  (ws + o); o += (size_t)H_ * EP_ * 2;        // 640 KB
    float* bias = (float*)(ws + o); o += (size_t)H_ * 4;              // 4 KB
    u16*   Whh  = (u16*)  (ws + o); o += (size_t)H_ * H_ * 2;         // 2 MB
    float* pre  = (float*)(ws + o); o += (size_t)B_ * T_ * H_ * 4;    // 128 MB
    u16*   h0   = (u16*)  (ws + o); o += (size_t)B_ * H_ * 2;         // 128 KB
    u16*   h1   = (u16*)  (ws + o); o += (size_t)B_ * H_ * 2;         // 128 KB
    float* hT   = (float*)(ws + o); o += (size_t)B_ * H_ * 4;         // 256 KB
    u32*   ctr  = (u32*)  (ws + o); o += 256;

    // zero h0+h1 (contiguous) and the barrier counter
    (void)hipMemsetAsync(h0, 0, (size_t)B_ * H_ * 2 * 2, stream);
    (void)hipMemsetAsync(ctr, 0, 256, stream);

    // allow 256 KB of dynamic LDS for the scan kernel (WGP has 320 KB)
    const int ldsBytes = 128 * H_ * (int)sizeof(u16);   // 262144
    (void)hipFuncSetAttribute((const void*)rnn_scan_kernel,
                              hipFuncAttributeMaxDynamicSharedMemorySize, ldsBytes);

    prep_weights_kernel<<<H_, EP_, 0, stream>>>(W_ih, W_proj, b_proj, b_ih, b_hh, Wc, bias);
    conv_whh_kernel<<<(H_ * H_) / 256, 256, 0, stream>>>(W_hh, Whh);
    gather_embed_kernel<<<B_ * T_, EP_, 0, stream>>>(x, emb, Xg);
    gemm_pre_kernel<<<4096, 256, 0, stream>>>(Xg, Wc, bias, pre);
    rnn_scan_kernel<<<8, 256, ldsBytes, stream>>>(pre, Whh, h0, h1, hT, ctr);
    fc_out_kernel<<<1, 1024, 0, stream>>>(hT, W_fc, b_fc, out);
}
